// BodyNetPart_52699248722029
// MI455X (gfx1250) — matrix operand order repaired and checked by reading, not compiled, not run on hardware
//
#include <hip/hip_runtime.h>
#include <math.h>

// ---------------------------------------------------------------------------
// CDNA5 WMMA types (wave32, v_wmma_f32_16x16x32_bf16)
// ---------------------------------------------------------------------------
typedef __bf16 bf16_t;
typedef bf16_t         v16bf __attribute__((ext_vector_type(16)));
typedef float          v8f   __attribute__((ext_vector_type(8)));
typedef unsigned int   v8u   __attribute__((ext_vector_type(8)));
typedef unsigned short u16_t;

__device__ __forceinline__ u16_t f2bfbits(float f) {
  union { float f; unsigned u; } a; a.f = f;
  return (u16_t)((a.u + 0x7FFFu + ((a.u >> 16) & 1u)) >> 16);   // RNE
}

// ---------------------------------------------------------------------------
// Weight repack: f32 [O,K] -> bf16 [O,Kp] zero-padded
// ---------------------------------------------------------------------------
__global__ void repack_bf16_kernel(const float* __restrict__ src,
                                   u16_t* __restrict__ dst,
                                   int K, int Kp, long total) {
  long t = (long)blockIdx.x * blockDim.x + threadIdx.x;
  if (t >= total) return;
  int k = (int)(t % Kp);
  long o = t / Kp;
  dst[t] = (k < K) ? f2bfbits(src[o * K + k]) : (u16_t)0;
}

// ---------------------------------------------------------------------------
// bf16 GEMM:  val = g[o]*sum_k A[m,k]*W[o,k] + bias[o]  (+optional ReLU)
// A: bf16 [M,lda], W: bf16 [O,ldw]; lda/ldw multiples of 32; Kp multiple of 32.
// Df (f32 [M,O], nullable), Db (bf16 [M,ldb], nullable, pad cols zeroed).
// One 16x16 tile per wave, 4 waves per block. Inner loop: 4x b128 load + WMMA.
// ---------------------------------------------------------------------------
__global__ void gemm_bf16_kernel(const u16_t* __restrict__ A, int lda,
                                 const u16_t* __restrict__ W, int ldw,
                                 const float* __restrict__ g,
                                 const float* __restrict__ bias,
                                 float* __restrict__ Df,
                                 u16_t* __restrict__ Db, int ldb,
                                 int M, int Kp, int O, int relu) {
  const int wave   = threadIdx.x >> 5;
  const int lane   = threadIdx.x & 31;
  const int mtile  = blockIdx.x * 4 + wave;
  const int tilesM = (M + 15) >> 4;
  if (mtile >= tilesM) return;                    // wave-uniform exit

  const int mbase = mtile * 16;
  const int obase = blockIdx.y * 16;
  const int row   = lane & 15;
  const int half  = lane >> 4;

  int am = mbase + row; if (am >= M) am = M - 1;  // clamp: loads unconditional
  int bo = obase + row; if (bo >= O) bo = O - 1;

  // A lane view: j0..3 -> k = k0+8h .. +7 ; j4..7 -> k = k0+16+8h .. +7
  const u16_t* Ap = A + (size_t)am * lda + 8 * half;
  // B lane view: j0..3 -> k = k0+8h .. +7 ; j4..7 -> k = k0+16+8h .. +7
  const u16_t* Wp = W + (size_t)bo * ldw + 8 * half;

  v8f acc = {0.f, 0.f, 0.f, 0.f, 0.f, 0.f, 0.f, 0.f};

  for (int k0 = 0; k0 < Kp; k0 += 32) {
    if (k0 + 32 < Kp) {
      __builtin_prefetch(Ap + k0 + 32, 0, 1);
      __builtin_prefetch(Wp + k0 + 32, 0, 1);
    }
    uint4 a0 = *(const uint4*)(Ap + k0);
    uint4 a1 = *(const uint4*)(Ap + k0 + 16);
    uint4 b0 = *(const uint4*)(Wp + k0);
    uint4 b1 = *(const uint4*)(Wp + k0 + 16);
    v8u au = {a0.x, a0.y, a0.z, a0.w, a1.x, a1.y, a1.z, a1.w};
    v8u bu = {b0.x, b0.y, b0.z, b0.w, b1.x, b1.y, b1.z, b1.w};
    acc = __builtin_amdgcn_wmma_f32_16x16x32_bf16(
        false, __builtin_bit_cast(v16bf, au),
        false, __builtin_bit_cast(v16bf, bu),
        (short)0, acc, false, false);
  }

  // C/D: lane -> col N = lane&15; VGPR r -> row M = r + 8*(lane>>4)
  const int oc = obase + row;
  float gv = 1.f, bv = 0.f;
  if (oc < O) {
    if (g)    gv = g[oc];
    if (bias) bv = bias[oc];
  }
#pragma unroll
  for (int r = 0; r < 8; ++r) {
    int m = mbase + r + 8 * half;
    if (m < M) {
      float v = acc[r] * gv + bv;
      if (relu) v = fmaxf(v, 0.f);
      if (Df && oc < O) Df[(size_t)m * O + oc] = v;
      if (Db) Db[(size_t)m * ldb + oc] = (oc < O) ? f2bfbits(v) : (u16_t)0;
    }
  }
}

// ---------------------------------------------------------------------------
// Farthest point sampling: one block per batch, dist kept in LDS.
// ---------------------------------------------------------------------------
#define FPS_T 256
__global__ void fps_kernel(const float* __restrict__ xyz, int* __restrict__ outIdx,
                           int N, int S) {
  __shared__ float dist[8192];
  __shared__ float rv[FPS_T];
  __shared__ int   ri[FPS_T];
  const int b = blockIdx.x, t = threadIdx.x;
  const float* x = xyz + (size_t)b * N * 3;
  for (int n = t; n < N; n += FPS_T) dist[n] = 1e10f;
  __syncthreads();
  int far = 0;
  for (int i = 0; i < S; ++i) {
    if (t == 0) outIdx[b * S + i] = far;
    float cx = x[far * 3 + 0], cy = x[far * 3 + 1], cz = x[far * 3 + 2];
    float bm = -1.f; int bi = 0x7FFFFFFF;
    for (int n = t; n < N; n += FPS_T) {
      float dx = x[n * 3 + 0] - cx, dy = x[n * 3 + 1] - cy, dz = x[n * 3 + 2] - cz;
      float d  = dx * dx + dy * dy + dz * dz;
      float dn = fminf(dist[n], d);
      dist[n] = dn;
      if (dn > bm) { bm = dn; bi = n; }
    }
    rv[t] = bm; ri[t] = bi;
    __syncthreads();
    for (int s2 = FPS_T / 2; s2 > 0; s2 >>= 1) {
      if (t < s2) {
        if (rv[t + s2] > rv[t] ||
            (rv[t + s2] == rv[t] && ri[t + s2] < ri[t])) {
          rv[t] = rv[t + s2]; ri[t] = ri[t + s2];
        }
      }
      __syncthreads();
    }
    far = ri[0];
    __syncthreads();
  }
}

__global__ void gather_xyz_kernel(const float* __restrict__ xyz,
                                  const int* __restrict__ fpsIdx,
                                  float* __restrict__ out,
                                  int N, int S, long total) {
  long t = (long)blockIdx.x * blockDim.x + threadIdx.x;
  if (t >= total) return;
  int j = (int)(t % 3);
  long r = t / 3;
  int s = (int)(r % S);
  int b = (int)(r / S);
  out[t] = xyz[((long)b * N + fpsIdx[b * S + s]) * 3 + j];
}

// ---------------------------------------------------------------------------
// Ball query: one wave per (b,s); first K ascending indices with d<=r^2,
// padded with first found (reference sentinel semantics). Wave32 ballot.
// ---------------------------------------------------------------------------
__global__ void ball_query_kernel(const float* __restrict__ xyz,
                                  const float* __restrict__ new_xyz,
                                  int* __restrict__ idx,
                                  int N, int S, int K, float r2) {
  const int bs = blockIdx.x;
  const int b = bs / S, s = bs % S;
  const int lane = threadIdx.x;
  const float* x = xyz + (size_t)b * N * 3;
  const float cx = new_xyz[((size_t)b * S + s) * 3 + 0];
  const float cy = new_xyz[((size_t)b * S + s) * 3 + 1];
  const float cz = new_xyz[((size_t)b * S + s) * 3 + 2];
  int* out = idx + ((size_t)b * S + s) * K;

  int count = 0, firstIdx = 0;
  bool haveFirst = false;
  for (int n0 = 0; n0 < N && count < K; n0 += 32) {
    int n = n0 + lane;
    bool ok = false;
    if (n < N) {
      float dx = x[n * 3 + 0] - cx, dy = x[n * 3 + 1] - cy, dz = x[n * 3 + 2] - cz;
      ok = (dx * dx + dy * dy + dz * dz) <= r2;
    }
    unsigned mask = __builtin_amdgcn_ballot_w32(ok);
    if (!haveFirst && mask) { firstIdx = n0 + __builtin_ctz(mask); haveFirst = true; }
    unsigned below = (lane == 0) ? 0u : (mask & ((1u << lane) - 1u));
    if (ok) {
      int pos = count + __builtin_popcount(below);
      if (pos < K) out[pos] = n;
    }
    count += __builtin_popcount(mask);
  }
  if (count > K) count = K;
  for (int k = count + lane; k < K; k += 32) out[k] = firstIdx;
}

// grouped input, bf16 + padded: out[m, c] m=(b*S+s)*K+k, stride Cp
__global__ void group_concat_bf16_kernel(const float* __restrict__ xyz,
                                         const float* __restrict__ pts,
                                         const float* __restrict__ new_xyz,
                                         const int* __restrict__ idx,
                                         u16_t* __restrict__ out,
                                         int N, int S, int K, int C, int Cp,
                                         long total) {
  long t = (long)blockIdx.x * blockDim.x + threadIdx.x;
  if (t >= total) return;
  int c = (int)(t % Cp);
  long r = t / Cp;
  int k = (int)(r % K);
  long r2 = r / K;
  int s = (int)(r2 % S);
  int b = (int)(r2 / S);
  float v = 0.f;
  if (c < C + 3) {
    int id = idx[((long)b * S + s) * K + k];
    if (c < C) {
      v = pts[((long)b * N + id) * C + c];
    } else {
      int j = c - C;
      v = xyz[((long)b * N + id) * 3 + j] - new_xyz[((long)b * S + s) * 3 + j];
    }
  }
  out[t] = f2bfbits(v);
}

// max-pool over K: in f32 [B,S,K,C] -> out[b,s,coff+c] (row width Ctot)
__global__ void maxpool_kernel(const float* __restrict__ in, float* __restrict__ out,
                               int S, int K, int C, int Ctot, int coff, long total) {
  long t = (long)blockIdx.x * blockDim.x + threadIdx.x;
  if (t >= total) return;
  int c = (int)(t % C);
  long r = t / C;
  int s = (int)(r % S);
  int b = (int)(r / S);
  const float* p = in + (((long)b * S + s) * K) * C + c;
  float m = p[0];
  for (int k = 1; k < K; ++k) m = fmaxf(m, p[(long)k * C]);
  out[((long)b * S + s) * Ctot + coff + c] = m;
}

// ---------------------------------------------------------------------------
// FP front-end: 3-NN inverse-distance interp + concat, bf16 padded output.
// ---------------------------------------------------------------------------
__global__ void fp_interp_concat_bf16_kernel(const float* __restrict__ xyz1,
                                             const float* __restrict__ xyz2,
                                             const float* __restrict__ pts1,
                                             const float* __restrict__ pts2,
                                             u16_t* __restrict__ out,
                                             int N1, int N2, int C1, int C2,
                                             int Cp, long total) {
  long t = (long)blockIdx.x * blockDim.x + threadIdx.x;
  if (t >= total) return;
  int n = (int)(t % N1);
  int b = (int)(t / N1);
  const float px = xyz1[((long)b * N1 + n) * 3 + 0];
  const float py = xyz1[((long)b * N1 + n) * 3 + 1];
  const float pz = xyz1[((long)b * N1 + n) * 3 + 2];
  float d0 = 1e30f, d1 = 1e30f, d2 = 1e30f;
  int i0 = 0, i1 = 0, i2 = 0;
  const float* x2 = xyz2 + (size_t)b * N2 * 3;
  for (int m = 0; m < N2; ++m) {
    float dx = x2[m * 3 + 0] - px, dy = x2[m * 3 + 1] - py, dz = x2[m * 3 + 2] - pz;
    float d = dx * dx + dy * dy + dz * dz;
    if (d < d0)      { d2 = d1; i2 = i1; d1 = d0; i1 = i0; d0 = d; i0 = m; }
    else if (d < d1) { d2 = d1; i2 = i1; d1 = d; i1 = m; }
    else if (d < d2) { d2 = d; i2 = m; }
  }
  float w0 = 1.f / (d0 + 1e-8f), w1 = 1.f / (d1 + 1e-8f), w2 = 1.f / (d2 + 1e-8f);
  float ws = w0 + w1 + w2;
  w0 /= ws; w1 /= ws; w2 /= ws;
  u16_t* o = out + ((long)b * N1 + n) * Cp;
  const float* p1r = pts1 + ((long)b * N1 + n) * C1;
  for (int c = 0; c < C1; ++c) o[c] = f2bfbits(p1r[c]);
  const float* q0 = pts2 + ((long)b * N2 + i0) * C2;
  const float* q1 = pts2 + ((long)b * N2 + i1) * C2;
  const float* q2 = pts2 + ((long)b * N2 + i2) * C2;
  for (int c = 0; c < C2; ++c)
    o[C1 + c] = f2bfbits(w0 * q0[c] + w1 * q1[c] + w2 * q2[c]);
  for (int c = C1 + C2; c < Cp; ++c) o[c] = 0;
}

// parts logits [B,N,P] -> parts out [B,P,N] + softmax over P stored [B,N,P]
__global__ void parts_softmax_kernel(const float* __restrict__ partsBN,
                                     float* __restrict__ partsOut,
                                     float* __restrict__ psm,
                                     int N, int P, long total) {
  long t = (long)blockIdx.x * blockDim.x + threadIdx.x;
  if (t >= total) return;
  int n = (int)(t % N);
  int b = (int)(t / N);
  const float* row = partsBN + ((long)b * N + n) * P;
  float mx = row[0];
  for (int p = 1; p < P; ++p) mx = fmaxf(mx, row[p]);
  float e[14], sum = 0.f;
  for (int p = 0; p < P; ++p) { e[p] = expf(row[p] - mx); sum += e[p]; }
  float inv = 1.f / sum;
  for (int p = 0; p < P; ++p) {
    partsOut[((long)b * P + p) * N + n] = row[p];
    psm[((long)b * N + n) * P + p] = e[p] * inv;
  }
}

// Fused pp head (avoids the 235MB y tensor): per (b,n) block, per part p:
// h = relu(p0 @ W1_p + b1_p); pred_{p,o} = h . W2[p,o] + b2[p,o];
// occ[cls] += pred * psm[(2p+o)%14],  cls = (2p+o)/14.
__global__ void pp_head_kernel(const float* __restrict__ p0,
                               const float* __restrict__ W1,
                               const float* __restrict__ b1,
                               const float* __restrict__ W2,
                               const float* __restrict__ b2,
                               const float* __restrict__ psm,
                               float* __restrict__ occOut, int N) {
  __shared__ float xr[128];
  __shared__ float r0[128];
  __shared__ float r1[128];
  const int bn = blockIdx.x;
  const int b = bn / N, n = bn % N;
  const int c = threadIdx.x;
  xr[c] = p0[(long)bn * 128 + c];
  __syncthreads();
  float occ0 = 0.f, occ1 = 0.f;
  for (int p = 0; p < 14; ++p) {
    const float* wrow = W1 + (long)(p * 128 + c) * 128;
    float acc = b1[p * 128 + c];
    for (int k = 0; k < 128; ++k) acc += xr[k] * wrow[k];
    float h = fmaxf(acc, 0.f);
    r0[c] = h * W2[(long)(p * 2 + 0) * 128 + c];
    r1[c] = h * W2[(long)(p * 2 + 1) * 128 + c];
    __syncthreads();
    for (int s2 = 64; s2 > 0; s2 >>= 1) {
      if (c < s2) { r0[c] += r0[c + s2]; r1[c] += r1[c + s2]; }
      __syncthreads();
    }
    if (c == 0) {
      float pr[2];
      pr[0] = r0[0] + b2[p * 2 + 0];
      pr[1] = r1[0] + b2[p * 2 + 1];
      for (int o = 0; o < 2; ++o) {
        int ch = p * 2 + o;
        float contrib = pr[o] * psm[(long)bn * 14 + (ch % 14)];
        if (ch / 14 == 0) occ0 += contrib; else occ1 += contrib;
      }
    }
    __syncthreads();
  }
  if (c == 0) {
    occOut[((long)b * 2 + 0) * N + n] = occ0;
    occOut[((long)b * 2 + 1) * N + n] = occ1;
  }
}

// ---------------------------------------------------------------------------
// Host orchestration
// ---------------------------------------------------------------------------
struct LayerH { int co; int wi; };

static inline long cdivl(long a, long b) { return (a + b - 1) / b; }
static inline int  pad32(int x) { return (x + 31) & ~31; }

static void run_gemm(hipStream_t st, void* const* d_in,
                     const u16_t* A, int lda, int wi, int O, int K,
                     u16_t* wscr, bool useG,
                     float* Df, u16_t* Db, int ldb, int M, int relu) {
  const float* Wsrc = (const float*)d_in[wi];
  const float* g = useG ? (const float*)d_in[wi + 1] : nullptr;
  const float* b = (const float*)d_in[wi + (useG ? 2 : 1)];
  int Kp = pad32(K);
  long tw = (long)O * Kp;
  repack_bf16_kernel<<<(unsigned)cdivl(tw, 256), 256, 0, st>>>(Wsrc, wscr, K, Kp, tw);
  int Ocov = Db ? pad32(O) : O;
  dim3 grid((unsigned)(((M + 15) / 16 + 3) / 4), (unsigned)((Ocov + 15) / 16));
  gemm_bf16_kernel<<<grid, 128, 0, st>>>(A, lda, wscr, Kp, g, b,
                                         Df, Db, ldb, M, Kp, O, relu);
}

static void mlp_chain(hipStream_t st, void* const* d_in,
                      const u16_t* A0, int lda0, int M, int K0,
                      const LayerH* ls, int nl,
                      u16_t* buf0, u16_t* buf1, u16_t* wscr,
                      float* destF, u16_t* destB, int destLdb) {
  const u16_t* cur = A0;
  int lda = lda0, K = K0, ping = 0;
  u16_t* bufs[2] = {buf0, buf1};
  for (int i = 0; i < nl; ++i) {
    bool last = (i == nl - 1);
    int co = ls[i].co, Opad = pad32(co);
    float* Df = last ? destF : nullptr;
    u16_t* Db; int ldb;
    if (last) { Db = destB; ldb = destLdb; }
    else {
      Db = bufs[ping];
      if (Db == cur) Db = bufs[ping ^ 1];
      ldb = Opad;
    }
    run_gemm(st, d_in, cur, lda, ls[i].wi, co, K, wscr, true, Df, Db, ldb, M, 1);
    cur = Db; lda = ldb; K = co; ping ^= 1;
  }
}

extern "C" void kernel_launch(void* const* d_in, const int* in_sizes, int n_in,
                              void* d_out, int out_size, void* d_ws, size_t ws_size,
                              hipStream_t stream) {
  (void)in_sizes; (void)n_in; (void)out_size; (void)ws_size;
  const int Bn = 4, Nn = 8192, P = 14;
  const float* inputs = (const float*)d_in[0];
  float* dout = (float*)d_out;

  // ---- workspace allocator (16B-aligned slabs) ----
  float* wsf = (float*)d_ws;
  size_t off = 0;
  auto alloc  = [&](size_t nf) { float* p = wsf + off; off += (nf + 7) & ~(size_t)7; return p; };
  auto allocB = [&](size_t nu) { return (u16_t*)alloc((nu + 1) / 2); };

  float* xyz1 = alloc((size_t)Bn * 1024 * 3);
  float* xyz2 = alloc((size_t)Bn * 256 * 3);
  float* xyz3 = alloc((size_t)Bn * 64 * 3);
  float* xyz4 = alloc((size_t)Bn * 16 * 3);
  float* pts1 = alloc((size_t)Bn * 1024 * 96);
  float* pts2 = alloc((size_t)Bn * 256 * 256);
  float* pts3 = alloc((size_t)Bn * 64 * 512);
  float* pts4 = alloc((size_t)Bn * 16 * 1024);
  float* p3   = alloc((size_t)Bn * 64 * 256);
  float* p2   = alloc((size_t)Bn * 256 * 256);
  float* p1   = alloc((size_t)Bn * 1024 * 128);
  float* p0   = alloc((size_t)Bn * Nn * 128);
  float* partsBN = alloc((size_t)Bn * Nn * P);
  float* psm     = alloc((size_t)Bn * Nn * P);
  float* actF    = alloc((size_t)4 * 1024 * 32 * 64);      // f32 final SA acts
  int* fpsIdx  = (int*)alloc((size_t)Bn * 1024);
  int* ballIdx = (int*)alloc((size_t)Bn * 1024 * 32);
  u16_t* groupedB = allocB((size_t)6 * 1024 * 1024);       // bf16 grouped/concat
  u16_t* actB0    = allocB((size_t)5 * 1024 * 1024);       // bf16 intermediates
  u16_t* actB1    = allocB((size_t)5 * 1024 * 1024);
  u16_t* p0b      = allocB((size_t)Bn * Nn * 128);
  u16_t* xb       = allocB((size_t)Bn * Nn * 128);
  u16_t* wscr     = allocB((size_t)640 * 1024);            // repacked weights

  // ---- SA stages ----
  struct BranchH { float r2; int K; LayerH L[3]; int coff; int cout; };
  struct StageH {
    int S, Nin, Cpts;
    const float* xyz_in; const float* pts_in;
    float* xyz_out; float* pts_out; int Ctot;
    BranchH br[2];
  };
  StageH stages[4] = {
    {1024, 8192, 3, inputs, inputs, xyz1, pts1, 96,
     {{0.05f * 0.05f, 16, {{16, 1}, {16, 4}, {32, 7}}, 0, 32},
      {0.10f * 0.10f, 32, {{32, 10}, {32, 13}, {64, 16}}, 32, 64}}},
    {256, 1024, 96, xyz1, pts1, xyz2, pts2, 256,
     {{0.10f * 0.10f, 16, {{64, 19}, {64, 22}, {128, 25}}, 0, 128},
      {0.20f * 0.20f, 32, {{64, 28}, {96, 31}, {128, 34}}, 128, 128}}},
    {64, 256, 256, xyz2, pts2, xyz3, pts3, 512,
     {{0.20f * 0.20f, 16, {{128, 37}, {196, 40}, {256, 43}}, 0, 256},
      {0.40f * 0.40f, 32, {{128, 46}, {196, 49}, {256, 52}}, 256, 256}}},
    {16, 64, 512, xyz3, pts3, xyz4, pts4, 1024,
     {{0.40f * 0.40f, 16, {{256, 55}, {256, 58}, {512, 61}}, 0, 512},
      {0.80f * 0.80f, 32, {{256, 64}, {384, 67}, {512, 70}}, 512, 512}}},
  };

  for (int st = 0; st < 4; ++st) {
    StageH& sp = stages[st];
    fps_kernel<<<Bn, FPS_T, 0, stream>>>(sp.xyz_in, fpsIdx, sp.Nin, sp.S);
    long tg = (long)Bn * sp.S * 3;
    gather_xyz_kernel<<<(unsigned)cdivl(tg, 256), 256, 0, stream>>>(
        sp.xyz_in, fpsIdx, sp.xyz_out, sp.Nin, sp.S, tg);
    for (int br = 0; br < 2; ++br) {
      BranchH& bs = sp.br[br];
      ball_query_kernel<<<Bn * sp.S, 32, 0, stream>>>(
          sp.xyz_in, sp.xyz_out, ballIdx, sp.Nin, sp.S, bs.K, bs.r2);
      int Cin = sp.Cpts + 3, Cp = pad32(Cin);
      long tc = (long)Bn * sp.S * bs.K * Cp;
      group_concat_bf16_kernel<<<(unsigned)cdivl(tc, 256), 256, 0, stream>>>(
          sp.xyz_in, sp.pts_in, sp.xyz_out, ballIdx, groupedB,
          sp.Nin, sp.S, bs.K, sp.Cpts, Cp, tc);
      int M = Bn * sp.S * bs.K;
      mlp_chain(stream, d_in, groupedB, Cp, M, Cin, bs.L, 3,
                actB0, actB1, wscr, actF, nullptr, 0);
      long tp = (long)Bn * sp.S * bs.cout;
      maxpool_kernel<<<(unsigned)cdivl(tp, 256), 256, 0, stream>>>(
          actF, sp.pts_out, sp.S, bs.K, bs.cout, sp.Ctot, bs.coff, tp);
    }
  }

  // ---- FP modules ----
  struct FPH {
    const float* xyz1; int N1; const float* xyz2; int N2;
    const float* pts1; int C1; const float* pts2; int C2;
    LayerH L[3]; int nl; float* dest; u16_t* destB; int destLdb;
  };
  FPH fpm[4] = {
    {xyz3, 64, xyz4, 16, pts3, 512, pts4, 1024,
     {{256, 73}, {256, 76}, {0, 0}}, 2, p3, nullptr, 0},
    {xyz2, 256, xyz3, 64, pts2, 256, p3, 256,
     {{256, 79}, {256, 82}, {0, 0}}, 2, p2, nullptr, 0},
    {xyz1, 1024, xyz2, 256, pts1, 96, p2, 256,
     {{256, 85}, {128, 88}, {0, 0}}, 2, p1, nullptr, 0},
    {inputs, Nn, xyz1, 1024, inputs, 3, p1, 128,
     {{128, 91}, {128, 94}, {128, 97}}, 3, p0, nullptr, 0},
  };
  fpm[3].destB = p0b; fpm[3].destLdb = 128;
  for (int f = 0; f < 4; ++f) {
    FPH& fp = fpm[f];
    int Cin = fp.C1 + fp.C2, Cp = pad32(Cin);
    long ti = (long)Bn * fp.N1;
    fp_interp_concat_bf16_kernel<<<(unsigned)cdivl(ti, 128), 128, 0, stream>>>(
        fp.xyz1, fp.xyz2, fp.pts1, fp.pts2, groupedB,
        fp.N1, fp.N2, fp.C1, fp.C2, Cp, ti);
    mlp_chain(stream, d_in, groupedB, Cp, Bn * fp.N1, Cin, fp.L, fp.nl,
              actB0, actB1, wscr, fp.dest, fp.destB, fp.destLdb);
  }

  // ---- Final heads ----
  int M = Bn * Nn;
  // x = relu(p0 @ W1^T * g1 + b1) -> bf16 xb
  run_gemm(stream, d_in, p0b, 128, 100, 128, 128, wscr, true,
           nullptr, xb, 128, M, 1);
  // parts logits = x @ W2^T + b2 -> f32 partsBN
  run_gemm(stream, d_in, xb, 128, 103, P, 128, wscr, false,
           partsBN, nullptr, 0, M, 0);
  long ts = (long)M;
  parts_softmax_kernel<<<(unsigned)cdivl(ts, 128), 128, 0, stream>>>(
      partsBN, dout, psm, Nn, P, ts);
  float* occOut = dout + (long)Bn * P * Nn;
  pp_head_kernel<<<M, 128, 0, stream>>>(
      p0, (const float*)d_in[105], (const float*)d_in[106],
      (const float*)d_in[107], (const float*)d_in[108], psm, occOut, Nn);
}
